// COLoRALinear_76527727280600
// MI455X (gfx1250) — matrix operand order repaired and checked by reading, not compiled, not run on hardware
//
#include <hip/hip_runtime.h>
#include <math.h>

// ---------------------------------------------------------------------------
// COLoRA fused linear for MI455X (gfx1250), wave32, WMMA bf16x3 split-precision
//   out = x @ W^T + b  +  cw*s * (x A_s^T) B_s^T  +  (1-cw)*s * sum_e r[b,e] (x A_e^T) B_e^T
// Folded as: P = scale(x @ Acat^T);  out = x @ W^T + P @ Bcat^T + b
// Software-pipelined: global fetch of K-tile kk+1 overlaps WMMA work on tile kk.
// ---------------------------------------------------------------------------

typedef unsigned int   u32;
typedef unsigned short u16;
typedef __attribute__((ext_vector_type(2)))  u32    u32x2;
typedef __attribute__((ext_vector_type(4)))  u32    u32x4;
typedef __attribute__((ext_vector_type(8)))  u32    u32x8;
typedef __attribute__((ext_vector_type(16))) __bf16 v16bf;
typedef __attribute__((ext_vector_type(8)))  float  v8f;

static_assert(sizeof(v16bf) == 32, "v16bf must be 32 bytes");

#define BN   4
#define SN   2048
#define DIN  2048
#define DOUT 2048
#define MTOT (BN * SN)          // 8192
#define KC   96                 // padded LoRA rank columns (72 -> 96)
#define LDSP 40                 // LDS row pitch in bf16 elems (32 data + 8 pad = 80B)
#define KSTEPS0 (DIN / 32)      // 64 base K-steps
#define KSTEPS1 (KC / 32)       // 3 LoRA epilogue K-steps

// ---- bf16 split helpers (truncation split: exact hi 16 bits, lo = residual) ----
static __device__ __forceinline__ u32 pack_bf16_trunc(float a, float b) {
  u32 ua = __builtin_bit_cast(u32, a);
  u32 ub = __builtin_bit_cast(u32, b);
  return (ua >> 16) | (ub & 0xFFFF0000u);
}
static __device__ __forceinline__ float bf16_hi_part(float a) {
  return __builtin_bit_cast(float, __builtin_bit_cast(u32, a) & 0xFFFF0000u);
}

static __device__ __forceinline__ void store_split(u16* H, u16* L, int row, int col, float4 v) {
  int base = row * LDSP + col;                      // col is a multiple of 4 -> 8B aligned
  u32x2 h = { pack_bf16_trunc(v.x, v.y), pack_bf16_trunc(v.z, v.w) };
  float lx = v.x - bf16_hi_part(v.x);
  float ly = v.y - bf16_hi_part(v.y);
  float lz = v.z - bf16_hi_part(v.z);
  float lw = v.w - bf16_hi_part(v.w);
  u32x2 l = { pack_bf16_trunc(lx, ly), pack_bf16_trunc(lz, lw) };
  *reinterpret_cast<u32x2*>(H + base) = h;
  *reinterpret_cast<u32x2*>(L + base) = l;
}

// Build a 16-element bf16 fragment from two 16-byte LDS chunks (both 16B aligned).
static __device__ __forceinline__ v16bf frag_from(const u16* p0, const u16* p1) {
  u32x4 a = *reinterpret_cast<const u32x4*>(p0);
  u32x4 b = *reinterpret_cast<const u32x4*>(p1);
  u32x8 r = { a[0], a[1], a[2], a[3], b[0], b[1], b[2], b[3] };
  return __builtin_bit_cast(v16bf, r);
}

// One wave's MMA work for a 32-K step: 2 m-tiles x NT n-tiles, bf16x3 split products.
// A-frag (16-bit A 16x32): lanes<16 -> k0..7 & k16..23 ; lanes>=16 -> k8..15 & k24..31
// B-frag (16-bit B 32x16): lanes<16 -> n=lane, k0..7 & k16..23 ; lanes>=16 -> n=lane-16, k8..15 & k24..31
template<int NT>
static __device__ __forceinline__ void wave_compute(const u16* AH, const u16* AL,
                                                    const u16* BH, const u16* BL,
                                                    int wm, int wn, int lane, v8f acc[2][NT]) {
  const int lr = lane & 15;
  const int hg = lane >> 4;
  v16bf ah[2], al[2];
#pragma unroll
  for (int im = 0; im < 2; ++im) {
    int base = (wm * 32 + im * 16 + lr) * LDSP + hg * 8;
    ah[im] = frag_from(AH + base, AH + base + 16);
    al[im] = frag_from(AL + base, AL + base + 16);
  }
  v16bf bh[NT], bl[NT];
#pragma unroll
  for (int in = 0; in < NT; ++in) {
    int base = (wn * (NT * 16) + in * 16 + lr) * LDSP + hg * 8;
    bh[in] = frag_from(BH + base, BH + base + 16);
    bl[in] = frag_from(BL + base, BL + base + 16);
  }
#pragma unroll
  for (int im = 0; im < 2; ++im)
#pragma unroll
    for (int in = 0; in < NT; ++in) {
      acc[im][in] = __builtin_amdgcn_wmma_f32_16x16x32_bf16(false, ah[im], false, bh[in],
                                                            (short)0, acc[im][in], false, false);
      acc[im][in] = __builtin_amdgcn_wmma_f32_16x16x32_bf16(false, ah[im], false, bl[in],
                                                            (short)0, acc[im][in], false, false);
      acc[im][in] = __builtin_amdgcn_wmma_f32_16x16x32_bf16(false, al[im], false, bh[in],
                                                            (short)0, acc[im][in], false, false);
    }
}

// ---------------------------------------------------------------------------
// Prep kernel 1: per-batch column mean of x  ->  xmean[B][DIN]
// ---------------------------------------------------------------------------
__global__ __launch_bounds__(256) void mean_kernel(const float* __restrict__ x,
                                                   float* __restrict__ xmean) {
  int b = blockIdx.x >> 3;
  int i = ((blockIdx.x & 7) << 8) + threadIdx.x;
  const float* p = x + ((size_t)b << 22) + i;       // b*S*DIN
  float s = 0.f;
  for (int si = 0; si < SN; ++si) s += p[(size_t)si << 11];
  xmean[(b << 11) + i] = s * (1.0f / (float)SN);
}

// ---------------------------------------------------------------------------
// Prep kernel 2: routing softmax + sigmoid(collab_w) -> per-column coefficients
//   coef[0]           = cw * SCALING                    (shared columns)
//   coef[1 + b*8 + e] = (1-cw) * SCALING * routing[b,e] (expert columns)
// ---------------------------------------------------------------------------
__global__ __launch_bounds__(256) void routing_kernel(const float* __restrict__ xmean,
                                                      const float* __restrict__ task_emb,
                                                      const float* __restrict__ collab_w,
                                                      float* __restrict__ coef) {
  __shared__ float logits[32];
  int tid = threadIdx.x, wave = tid >> 5, lane = tid & 31;
  for (int p = wave; p < 32; p += 8) {
    int b = p >> 3, e = p & 7;
    const float* xm = xmean + (b << 11);
    const float* te = task_emb + (e << 11);
    float s = 0.f;
    for (int i = lane; i < DIN; i += 32) s += xm[i] * te[i];
    for (int off = 16; off; off >>= 1) s += __shfl_down(s, off);
    if (lane == 0) logits[p] = s;
  }
  __syncthreads();
  float cw = 1.0f / (1.0f + __expf(-collab_w[0]));
  if (tid == 0) coef[0] = cw * 2.0f;                // SCALING = 2
  if (tid < 4) {
    int b = tid;
    float mx = -1e30f;
    for (int e = 0; e < 8; ++e) mx = fmaxf(mx, logits[b * 8 + e]);
    float ex[8], sum = 0.f;
    for (int e = 0; e < 8; ++e) { ex[e] = __expf(logits[b * 8 + e] - mx); sum += ex[e]; }
    float k = (1.0f - cw) * 2.0f / sum;
    for (int e = 0; e < 8; ++e) coef[1 + b * 8 + e] = ex[e] * k;
  }
}

// ---------------------------------------------------------------------------
// Prep kernel 3: Bcat^T [DOUT][KC] : c<8 -> shared_B[o][c]; 8..71 -> expert_B[e][o][r]; pad 0
// ---------------------------------------------------------------------------
__global__ __launch_bounds__(256) void bcat_kernel(const float* __restrict__ shared_B,
                                                   const float* __restrict__ expert_B,
                                                   float* __restrict__ bcat) {
  int idx = blockIdx.x * 256 + threadIdx.x;
  if (idx >= DOUT * KC) return;
  int o = idx / KC, c = idx % KC;
  float v = 0.f;
  if (c < 8)       v = shared_B[o * 8 + c];
  else if (c < 72) { int q = c - 8; v = expert_B[(q >> 3) * (DOUT * 8) + o * 8 + (q & 7)]; }
  bcat[idx] = v;
}

// ---------------------------------------------------------------------------
// Register fetch helpers (software pipeline stage): 8 threads per row, float4 each
// ---------------------------------------------------------------------------
static __device__ __forceinline__ void fetch_main(int kk,
                                                  const float* __restrict__ x,
                                                  const float* __restrict__ W,
                                                  const float* __restrict__ P,
                                                  const float* __restrict__ bcat,
                                                  int m0, int n0, int tid,
                                                  float4 ra[4], float4 rb[4]) {
  const float* ga;
  const float* gb;
  long pitch;
  if (kk < KSTEPS0) {
    ga = x + (size_t)m0 * DIN + kk * 32;
    gb = W + (size_t)n0 * DIN + kk * 32;
    pitch = DIN;
  } else {
    int k2 = kk - KSTEPS0;
    ga = P    + (size_t)m0 * KC + k2 * 32;
    gb = bcat + (size_t)n0 * KC + k2 * 32;
    pitch = KC;
  }
  const int col = (tid & 7) * 4;
  const int r0  = tid >> 3;
#pragma unroll
  for (int p = 0; p < 4; ++p) {
    const float* sa = ga + (size_t)(p * 32 + r0) * pitch + col;
    const float* sb = gb + (size_t)(p * 32 + r0) * pitch + col;
    __builtin_prefetch(sa + 64, 0, 1);              // two K-steps ahead (speculative)
    __builtin_prefetch(sb + 64, 0, 1);
    ra[p] = *reinterpret_cast<const float4*>(sa);
    rb[p] = *reinterpret_cast<const float4*>(sb);
  }
}

static __device__ __forceinline__ void fetch_lora(int kk,
                                                  const float* __restrict__ x,
                                                  const float* __restrict__ shared_A,
                                                  const float* __restrict__ expert_A,
                                                  int m0, int tid,
                                                  float4 ra[4], float4 rb[3]) {
  const int col = (tid & 7) * 4;
  const int r0  = tid >> 3;
  const float* ga = x + (size_t)m0 * DIN + kk * 32;
#pragma unroll
  for (int p = 0; p < 4; ++p) {
    const float* sa = ga + (size_t)(p * 32 + r0) * DIN + col;
    __builtin_prefetch(sa + 64, 0, 1);
    ra[p] = *reinterpret_cast<const float4*>(sa);
  }
  const int k = kk * 32 + col;
#pragma unroll
  for (int p = 0; p < 3; ++p) {
    int row = p * 32 + r0;                          // 0..95, rows >= 72 are zero pad
    float4 v = make_float4(0.f, 0.f, 0.f, 0.f);
    if (row < 8)       v = *reinterpret_cast<const float4*>(shared_A + (size_t)row * DIN + k);
    else if (row < 72) v = *reinterpret_cast<const float4*>(expert_A + (size_t)(row - 8) * DIN + k);
    rb[p] = v;
  }
}

// ---------------------------------------------------------------------------
// LoRA projection GEMM: P[MTOT][KC] = scale( x[MTOT][DIN] @ Acat^T ), Acat = [72][DIN]
// Block tile 128(m) x 96(c), 8 waves (4m x 2n), 2x3 16x16 tiles/wave, K=2048
// ---------------------------------------------------------------------------
__global__ __launch_bounds__(256) void colora_lora(const float* __restrict__ x,
                                                   const float* __restrict__ shared_A,
                                                   const float* __restrict__ expert_A,
                                                   const float* __restrict__ coef,
                                                   float* __restrict__ P) {
  __shared__ u16 AH[128 * LDSP], AL[128 * LDSP];
  __shared__ u16 BH[96  * LDSP], BL[96  * LDSP];
  const int tid  = threadIdx.x;
  const int m0   = blockIdx.x * 128;
  const int bidx = m0 >> 11;                        // batch index (128 | 2048)
  const int wave = tid >> 5, lane = tid & 31;
  const int wm = wave & 3, wn = wave >> 2;
  const int col = (tid & 7) * 4;
  const int r0  = tid >> 3;

  v8f acc[2][3];
  v8f z = {0.f, 0.f, 0.f, 0.f, 0.f, 0.f, 0.f, 0.f};
#pragma unroll
  for (int i = 0; i < 2; ++i)
#pragma unroll
    for (int j = 0; j < 3; ++j) acc[i][j] = z;

  float4 ra[4], rb[3];
  fetch_lora(0, x, shared_A, expert_A, m0, tid, ra, rb);

  for (int kk = 0; kk < KSTEPS0; ++kk) {
#pragma unroll
    for (int p = 0; p < 4; ++p) store_split(AH, AL, p * 32 + r0, col, ra[p]);
#pragma unroll
    for (int p = 0; p < 3; ++p) store_split(BH, BL, p * 32 + r0, col, rb[p]);
    __syncthreads();
    if (kk + 1 < KSTEPS0)
      fetch_lora(kk + 1, x, shared_A, expert_A, m0, tid, ra, rb);  // overlaps compute
    wave_compute<3>(AH, AL, BH, BL, wm, wn, lane, acc);
    __syncthreads();
  }

  const float coefS = coef[0];
#pragma unroll
  for (int im = 0; im < 2; ++im) {
    int mbase = m0 + wm * 32 + im * 16 + (lane >> 4) * 8;
#pragma unroll
    for (int in = 0; in < 3; ++in) {
      int c = wn * 48 + in * 16 + (lane & 15);
      float cf = (c < 8) ? coefS : ((c < 72) ? coef[1 + bidx * 8 + ((c - 8) >> 3)] : 0.0f);
#pragma unroll
      for (int v = 0; v < 8; ++v)
        P[(size_t)(mbase + v) * KC + c] = acc[im][in][v] * cf;
    }
  }
}

// ---------------------------------------------------------------------------
// Main GEMM: out = x @ W^T (K=2048)  +  P @ Bcat^T (K'=96 epilogue)  +  bias
// Block tile 128x128, 8 waves (4m x 2n), 2x4 16x16 tiles/wave, pipelined loop
// ---------------------------------------------------------------------------
__global__ __launch_bounds__(256) void colora_main(const float* __restrict__ x,
                                                   const float* __restrict__ W,
                                                   const float* __restrict__ bias,
                                                   const float* __restrict__ P,
                                                   const float* __restrict__ bcat,
                                                   float* __restrict__ out) {
  __shared__ u16 AH[128 * LDSP], AL[128 * LDSP];
  __shared__ u16 BH[128 * LDSP], BL[128 * LDSP];
  const int tid = threadIdx.x;
  const int n0 = blockIdx.x * 128;
  const int m0 = blockIdx.y * 128;
  const int wave = tid >> 5, lane = tid & 31;
  const int wm = wave & 3, wn = wave >> 2;
  const int col = (tid & 7) * 4;
  const int r0  = tid >> 3;

  v8f acc[2][4];
  v8f z = {0.f, 0.f, 0.f, 0.f, 0.f, 0.f, 0.f, 0.f};
#pragma unroll
  for (int i = 0; i < 2; ++i)
#pragma unroll
    for (int j = 0; j < 4; ++j) acc[i][j] = z;

  const int STEPS = KSTEPS0 + KSTEPS1;              // 67 unified K-steps
  float4 ra[4], rb[4];
  fetch_main(0, x, W, P, bcat, m0, n0, tid, ra, rb);

  for (int kk = 0; kk < STEPS; ++kk) {
#pragma unroll
    for (int p = 0; p < 4; ++p) {
      store_split(AH, AL, p * 32 + r0, col, ra[p]);
      store_split(BH, BL, p * 32 + r0, col, rb[p]);
    }
    __syncthreads();
    if (kk + 1 < STEPS)
      fetch_main(kk + 1, x, W, P, bcat, m0, n0, tid, ra, rb);      // overlaps compute
    wave_compute<4>(AH, AL, BH, BL, wm, wn, lane, acc);
    __syncthreads();
  }

#pragma unroll
  for (int im = 0; im < 2; ++im) {
    int mbase = m0 + wm * 32 + im * 16 + (lane >> 4) * 8;
#pragma unroll
    for (int in = 0; in < 4; ++in) {
      int n = n0 + wn * 64 + in * 16 + (lane & 15);
      float bv = bias[n];
#pragma unroll
      for (int v = 0; v < 8; ++v)
        out[(size_t)(mbase + v) * DOUT + n] = acc[im][in][v] + bv;
    }
  }
}

// ---------------------------------------------------------------------------
extern "C" void kernel_launch(void* const* d_in, const int* in_sizes, int n_in,
                              void* d_out, int out_size, void* d_ws, size_t ws_size,
                              hipStream_t stream) {
  (void)in_sizes; (void)n_in; (void)out_size; (void)ws_size;
  const float* x        = (const float*)d_in[0];
  const float* base_W   = (const float*)d_in[1];
  const float* base_b   = (const float*)d_in[2];
  const float* shared_A = (const float*)d_in[3];
  const float* shared_B = (const float*)d_in[4];
  const float* expert_A = (const float*)d_in[5];
  const float* expert_B = (const float*)d_in[6];
  const float* task_emb = (const float*)d_in[7];
  const float* collab_w = (const float*)d_in[8];
  float* out = (float*)d_out;

  float* ws    = (float*)d_ws;
  float* xmean = ws;                               // 8192 floats
  float* coef  = ws + 8192;                        // 64 floats (1 + 32 used)
  float* bcat  = ws + 8192 + 64;                   // DOUT*KC = 196608 floats
  float* P     = ws + 8192 + 64 + DOUT * KC;       // MTOT*KC = 786432 floats  (~4 MB total)

  mean_kernel   <<<32, 256, 0, stream>>>(x, xmean);
  routing_kernel<<<1, 256, 0, stream>>>(xmean, task_emb, collab_w, coef);
  bcat_kernel   <<<(DOUT * KC + 255) / 256, 256, 0, stream>>>(shared_B, expert_B, bcat);
  colora_lora   <<<MTOT / 128, 256, 0, stream>>>(x, shared_A, expert_A, coef, P);
  colora_main   <<<dim3(DOUT / 128, MTOT / 128), 256, 0, stream>>>(x, base_W, base_b, P, bcat, out);
}
